// GNNEncoder_33741263077798
// MI455X (gfx1250) — matrix operand order repaired and checked by reading, not compiled, not run on hardware
//
#include <hip/hip_runtime.h>
#include <hip/hip_bf16.h>

// ---------------------------------------------------------------------------
// GatedGCN encoder for MI455X (gfx1250).  B=2, V=256, H=128, L=2, T=64.
// All H->H linears use v_wmma_f32_16x16x32_bf16 (bf16 in, f32 accum).
// Memory-bound: this revision fuses the 4 gating GEMMs + gate/message/j-sum
// into one kernel operating directly on WMMA accumulator layout, fuses the
// two e-path LayerNorms, and folds the plo residual into the GEMM epilogue.
// Edge-sized (64MB) passes: ~26/layer -> ~12/layer  (~167us -> ~95us @23.3TB/s).
//
// Param flattening assumption (jax.tree_leaves: dict keys sorted, lists in
// order, tuples (w,b) in order):
//   0 label  1 edge_weights  2 adj_matrix(int)  3 xt  4 t(int)
//   5/6 edge_embed  7/8 edge_pos_embed  9/10 ew_embed  11/12 ew_pos_embed
//   13 + 28*l: A(0,1) B(2,3) C(4,5) D(6,7) U(8,9) Uew(10,11) V(12,13)
//              Vew(14,15) ln_e(16,17) ln_ew(18,19) ln_h(20,21)
//              plo_lin(22,23) plo_ln(24,25) time_proj(26,27)
//   69/70 node_embed  71/72 out_conv  73/74 out_gn  75/76 pos_embed
//   77/78 time1  79/80 time2
// ---------------------------------------------------------------------------

typedef __attribute__((ext_vector_type(16))) __bf16 v16bf;
typedef __attribute__((ext_vector_type(8)))  float  v8f;

#define BB   2
#define VV   256
#define HH   128
#define TT   64
#define BV   (BB*VV)          // 512
#define PIX  (BB*VV*VV)       // 131072
#define PPB  (VV*VV)          // 65536 pixels per batch
#define EN   ((long)PIX*HH)   // edge-sized element count

// ---------------------------------------------------------------------------
// Generic WMMA GEMM: Y[M,128] = (res?) + X[M,128] @ W[128,128]^T + bias
// block = 256 thr = 8 waves; block tile = 128x128; M % 128 == 0.
// ---------------------------------------------------------------------------
__global__ __launch_bounds__(256) void gemm128_wmma(
    const float* __restrict__ X, const float* __restrict__ W,
    const float* __restrict__ bias, const float* __restrict__ res,
    float* __restrict__ Y, int M)
{
  __shared__ __bf16 Xs[128*128];
  __shared__ __bf16 Ws[128*128];
  const int  tid = threadIdx.x;
  const long m0  = (long)blockIdx.x * 128;

  for (int i = tid; i < 128*128/4; i += 256) {
    float4 xv = ((const float4*)(X + m0*128))[i];
    Xs[i*4+0] = (__bf16)xv.x; Xs[i*4+1] = (__bf16)xv.y;
    Xs[i*4+2] = (__bf16)xv.z; Xs[i*4+3] = (__bf16)xv.w;
    float4 wv = ((const float4*)W)[i];
    Ws[i*4+0] = (__bf16)wv.x; Ws[i*4+1] = (__bf16)wv.y;
    Ws[i*4+2] = (__bf16)wv.z; Ws[i*4+3] = (__bf16)wv.w;
  }
  __syncthreads();

  const int lane  = tid & 31;
  const int lrow  = lane & 15;
  const int hiA   = (lane < 16) ? 0 : 16;  // A-frag K offset
  const int hiB   = (lane < 16) ? 0 : 16;  // B-frag K offset
  const int ncol0 = (tid >> 5) * 16;

  v8f acc[8];
  #pragma unroll
  for (int mt = 0; mt < 8; ++mt) acc[mt] = (v8f){};

  #pragma unroll
  for (int kc = 0; kc < 4; ++kc) {
    const int k0 = kc * 32;
    v16bf bfrag;
    {
      const __bf16* wp = &Ws[(ncol0 + lrow)*128 + k0 + hiB];
      #pragma unroll
      for (int i = 0; i < 16; ++i) bfrag[i] = wp[i];
    }
    #pragma unroll
    for (int mt = 0; mt < 8; ++mt) {
      const __bf16* xp = &Xs[(mt*16 + lrow)*128 + k0 + hiA];
      v16bf afrag;
      #pragma unroll
      for (int i = 0; i < 8; ++i) { afrag[i] = xp[i]; afrag[8+i] = xp[8+i]; }
      acc[mt] = __builtin_amdgcn_wmma_f32_16x16x32_bf16(
          false, afrag, false, bfrag, (short)0, acc[mt], false, false);
    }
  }

  const float bn = bias[ncol0 + lrow];
  const int   ro = (lane < 16) ? 0 : 8;
  #pragma unroll
  for (int mt = 0; mt < 8; ++mt)
    #pragma unroll
    for (int r = 0; r < 8; ++r) {
      long idx = (m0 + mt*16 + r + ro)*128 + ncol0 + lrow;
      float v = acc[mt][r] + bn;
      if (res) v += res[idx];
      Y[idx] = v;
    }
}

// ---------------------------------------------------------------------------
// Fused gate-GEMM: computes Ce = e@C^T, Dew = ew@D^T, Uew = ew@Uew^T,
// Vew = ew@Vew^T for a 64-pixel tile, then the gating stage directly on the
// accumulator layout:
//   ewn = (Ah[j]+Bh[i]+Dew)*g ; en = ewn + Ce*g ; gate = sigmoid(en)
//   t0 = en ; t1 = ewn + Uew ; h_n[b,i,:] += sum_j gate*(Vh[j]+Vew)*g
// Block tile = 64 rows x 128 cols (rows m0..m0+63 share (b,i) since 64|256).
// ---------------------------------------------------------------------------
__global__ __launch_bounds__(256) void gemm_gate(
    const float* __restrict__ e,  const float* __restrict__ ew,
    const float* __restrict__ Wc, const float* __restrict__ bc,
    const float* __restrict__ Wd, const float* __restrict__ bd,
    const float* __restrict__ Wu, const float* __restrict__ bu,
    const float* __restrict__ Wv, const float* __restrict__ bv,
    const int*   __restrict__ adj,
    const float* __restrict__ Ah, const float* __restrict__ Bh,
    const float* __restrict__ Vh,
    float* __restrict__ t0, float* __restrict__ t1,
    float* __restrict__ h_n)
{
  __shared__ __bf16 Es[64*128];
  __shared__ __bf16 EWs[64*128];
  const int  tid = threadIdx.x;
  const long m0  = (long)blockIdx.x * 64;
  const int  bi  = (int)(m0 >> 8);     // b*V + i (constant over the tile)
  const int  b   = (int)(m0 >> 16);

  for (int i = tid; i < 64*128/4; i += 256) {
    float4 xv = ((const float4*)(e + m0*128))[i];
    Es[i*4+0] = (__bf16)xv.x; Es[i*4+1] = (__bf16)xv.y;
    Es[i*4+2] = (__bf16)xv.z; Es[i*4+3] = (__bf16)xv.w;
    float4 wv = ((const float4*)(ew + m0*128))[i];
    EWs[i*4+0] = (__bf16)wv.x; EWs[i*4+1] = (__bf16)wv.y;
    EWs[i*4+2] = (__bf16)wv.z; EWs[i*4+3] = (__bf16)wv.w;
  }
  __syncthreads();

  const int lane  = tid & 31;
  const int lrow  = lane & 15;
  const int hiA   = (lane < 16) ? 0 : 16;
  const int hiB   = (lane < 16) ? 0 : 16;
  const int ncol0 = (tid >> 5) * 16;
  const int n     = ncol0 + lrow;          // this lane's output channel

  v8f aC[4], aD[4], aU[4], aV[4];
  #pragma unroll
  for (int mt = 0; mt < 4; ++mt) { aC[mt]=(v8f){}; aD[mt]=(v8f){}; aU[mt]=(v8f){}; aV[mt]=(v8f){}; }

  #pragma unroll
  for (int kc = 0; kc < 4; ++kc) {
    const int k0 = kc * 32;
    // weight fragments straight from global (64KB each -> L2 resident)
    v16bf fC, fD, fU, fV;
    {
      const long wo = (long)n*128 + k0 + hiB;
      const float4* pc = (const float4*)(Wc + wo);
      const float4* pd = (const float4*)(Wd + wo);
      const float4* pu = (const float4*)(Wu + wo);
      const float4* pv = (const float4*)(Wv + wo);
      #pragma unroll
      for (int q = 0; q < 4; ++q) {
        float4 c = pc[q], d = pd[q], u = pu[q], v = pv[q];
        fC[q*4+0]=(__bf16)c.x; fC[q*4+1]=(__bf16)c.y; fC[q*4+2]=(__bf16)c.z; fC[q*4+3]=(__bf16)c.w;
        fD[q*4+0]=(__bf16)d.x; fD[q*4+1]=(__bf16)d.y; fD[q*4+2]=(__bf16)d.z; fD[q*4+3]=(__bf16)d.w;
        fU[q*4+0]=(__bf16)u.x; fU[q*4+1]=(__bf16)u.y; fU[q*4+2]=(__bf16)u.z; fU[q*4+3]=(__bf16)u.w;
        fV[q*4+0]=(__bf16)v.x; fV[q*4+1]=(__bf16)v.y; fV[q*4+2]=(__bf16)v.z; fV[q*4+3]=(__bf16)v.w;
      }
    }
    #pragma unroll
    for (int mt = 0; mt < 4; ++mt) {
      v16bf aE, aW;
      const __bf16* pe = &Es [(mt*16 + lrow)*128 + k0 + hiA];
      const __bf16* pw = &EWs[(mt*16 + lrow)*128 + k0 + hiA];
      #pragma unroll
      for (int i = 0; i < 8; ++i) { aE[i]=pe[i]; aE[8+i]=pe[8+i]; aW[i]=pw[i]; aW[8+i]=pw[8+i]; }
      aC[mt] = __builtin_amdgcn_wmma_f32_16x16x32_bf16(false, aE, false, fC, (short)0, aC[mt], false, false);
      aD[mt] = __builtin_amdgcn_wmma_f32_16x16x32_bf16(false, aW, false, fD, (short)0, aD[mt], false, false);
      aU[mt] = __builtin_amdgcn_wmma_f32_16x16x32_bf16(false, aW, false, fU, (short)0, aU[mt], false, false);
      aV[mt] = __builtin_amdgcn_wmma_f32_16x16x32_bf16(false, aW, false, fV, (short)0, aV[mt], false, false);
    }
  }

  // gating epilogue on accumulator layout
  const float cbn = bc[n], dbn = bd[n], ubn = bu[n], vbn = bv[n];
  const float bh  = Bh[bi*HH + n];
  const int   ro  = (lane < 16) ? 0 : 8;
  float hacc = 0.f;
  #pragma unroll
  for (int mt = 0; mt < 4; ++mt) {
    #pragma unroll
    for (int r = 0; r < 8; ++r) {
      const long m   = m0 + mt*16 + r + ro;       // pixel index
      const int  j   = (int)(m & 255);
      const long idx = m*128 + n;
      const float g  = (float)adj[bi*VV + j];
      const float ah = Ah[(b*VV + j)*HH + n];
      const float vh = Vh[(b*VV + j)*HH + n];
      const float ce  = aC[mt][r] + cbn;
      const float dew = aD[mt][r] + dbn;
      const float uew = aU[mt][r] + ubn;
      const float vew = aV[mt][r] + vbn;
      const float ewn = (ah + bh + dew) * g;
      const float en  = ewn + ce * g;
      const float gate = 1.f / (1.f + __expf(-en));
      hacc += gate * (vh + vew) * g;
      t0[idx] = en;
      t1[idx] = ewn + uew;
    }
  }
  // lanes l and l^16 share channel n and cover complementary rows
  hacc += __shfl_xor(hacc, 16, 32);
  if (lane < 16) atomicAdd(&h_n[bi*HH + n], hacc);
}

// y[p,h] = s[p]*w[h] + b[h]
__global__ void k_posembed(const float* __restrict__ s, const float* __restrict__ w,
                           const float* __restrict__ b, float* __restrict__ y, long n)
{
  long i = (long)blockIdx.x * blockDim.x + threadIdx.x;
  if (i >= n) return;
  long p = i >> 7; int h = (int)(i & 127);
  y[i] = s[p] * w[h] + b[h];
}

__global__ void k_copy(float* __restrict__ dst, const float* __restrict__ src, long n)
{
  long i = (long)blockIdx.x * blockDim.x + threadIdx.x;
  if (i < n) dst[i] = src[i];
}

// timestep embedding + time MLP (1 block, 128 threads)
__global__ __launch_bounds__(128) void k_time(
    const int* __restrict__ t,
    const float* __restrict__ W1, const float* __restrict__ b1,
    const float* __restrict__ W2, const float* __restrict__ b2,
    float* __restrict__ tm2)
{
  __shared__ float temb[BB][HH];
  __shared__ float tm1[BB][TT];
  const int tid = threadIdx.x;
  for (int b = 0; b < BB; ++b) {
    float tf   = (float)t[b];
    float freq = __expf(-logf(10000.f) * (float)(tid & 63) / 64.f);
    float arg  = tf * freq;
    temb[b][tid] = (tid < 64) ? cosf(arg) : sinf(arg);
  }
  __syncthreads();
  { int b = tid >> 6, o = tid & 63;
    float s = b1[o];
    for (int k = 0; k < HH; ++k) s += temb[b][k] * W1[o*HH + k];
    tm1[b][o] = fmaxf(s, 0.f); }
  __syncthreads();
  { int b = tid >> 6, o = tid & 63;
    float s = b2[o];
    for (int k = 0; k < TT; ++k) s += tm1[b][k] * W2[o*TT + k];
    tm2[b*TT + o] = s; }
}

// tproj[b,h] = relu(tm2[b]) @ Wp^T + bp  (1 block, 256 threads)
__global__ __launch_bounds__(256) void k_tproj(
    const float* __restrict__ tm2, const float* __restrict__ Wp,
    const float* __restrict__ bp, float* __restrict__ tp)
{
  int b = threadIdx.x >> 7, h = threadIdx.x & 127;
  float s = bp[h];
  for (int k = 0; k < TT; ++k) s += fmaxf(tm2[b*TT + k], 0.f) * Wp[h*TT + k];
  tp[b*HH + h] = s;
}

// Per-row LayerNorm: out = res + relu(ln(in))
__global__ __launch_bounds__(128) void k_ln_res(
    const float* __restrict__ in, const float* __restrict__ gamma,
    const float* __restrict__ beta, const float* __restrict__ res,
    float* __restrict__ out)
{
  __shared__ float red[128];
  const long row = blockIdx.x;
  const int  h   = threadIdx.x;
  float v = in[row*HH + h];
  red[h] = v; __syncthreads();
  for (int s = 64; s > 0; s >>= 1) { if (h < s) red[h] += red[h+s]; __syncthreads(); }
  float mean = red[0] * (1.f/128.f); __syncthreads();
  float d = v - mean;
  red[h] = d*d; __syncthreads();
  for (int s = 64; s > 0; s >>= 1) { if (h < s) red[h] += red[h+s]; __syncthreads(); }
  float var = red[0] * (1.f/128.f);
  float y = d * rsqrtf(var + 1e-5f) * gamma[h] + beta[h];
  out[row*HH + h] = res[row*HH + h] + fmaxf(y, 0.f);
}

// Dual LayerNorm (e path): y1 = relu(ln_e(in)) + tproj[b];
// out = silu(plo_ln(y1)).  Only the silu result hits memory.
__global__ __launch_bounds__(128) void k_ln2(
    const float* __restrict__ in,
    const float* __restrict__ g1, const float* __restrict__ b1,
    const float* __restrict__ tproj,
    const float* __restrict__ g2, const float* __restrict__ b2,
    float* __restrict__ out, int pixPerB)
{
  __shared__ float red[128];
  const long row = blockIdx.x;
  const int  h   = threadIdx.x;
  const int  b   = (int)(row / pixPerB);

  float v = in[row*HH + h];
  red[h] = v; __syncthreads();
  for (int s = 64; s > 0; s >>= 1) { if (h < s) red[h] += red[h+s]; __syncthreads(); }
  float mean = red[0] * (1.f/128.f); __syncthreads();
  float d = v - mean;
  red[h] = d*d; __syncthreads();
  for (int s = 64; s > 0; s >>= 1) { if (h < s) red[h] += red[h+s]; __syncthreads(); }
  float var = red[0] * (1.f/128.f);
  float y1 = fmaxf(d * rsqrtf(var + 1e-5f) * g1[h] + b1[h], 0.f) + tproj[b*HH + h];
  __syncthreads();

  red[h] = y1; __syncthreads();
  for (int s = 64; s > 0; s >>= 1) { if (h < s) red[h] += red[h+s]; __syncthreads(); }
  float mean2 = red[0] * (1.f/128.f); __syncthreads();
  float d2 = y1 - mean2;
  red[h] = d2*d2; __syncthreads();
  for (int s = 64; s > 0; s >>= 1) { if (h < s) red[h] += red[h+s]; __syncthreads(); }
  float var2 = red[0] * (1.f/128.f);
  float y2 = d2 * rsqrtf(var2 + 1e-5f) * g2[h] + b2[h];
  out[row*HH + h] = y2 / (1.f + __expf(-y2));
}

// GroupNorm stats: block = (b,g), mean/rstd over 4 channels x 65536 px
__global__ __launch_bounds__(256) void k_gnstats(const float* __restrict__ e,
                                                 float* __restrict__ stats)
{
  __shared__ float s1[256], s2[256];
  const int bg = blockIdx.x;
  const int b  = bg >> 5, g = bg & 31;
  const long ebase = (long)b * PPB * HH;
  float a = 0.f, q = 0.f;
  for (int p = threadIdx.x; p < PPB; p += 256) {
    const float4 v = *(const float4*)(e + ebase + (long)p*HH + g*4);
    a += v.x + v.y + v.z + v.w;
    q += v.x*v.x + v.y*v.y + v.z*v.z + v.w*v.w;
  }
  s1[threadIdx.x] = a; s2[threadIdx.x] = q; __syncthreads();
  for (int s = 128; s > 0; s >>= 1) {
    if (threadIdx.x < s) { s1[threadIdx.x] += s1[threadIdx.x+s]; s2[threadIdx.x] += s2[threadIdx.x+s]; }
    __syncthreads();
  }
  if (threadIdx.x == 0) {
    float nn = (float)PPB * 4.f;
    float mean = s1[0] / nn;
    float var  = s2[0] / nn - mean*mean;
    stats[bg*2]   = mean;
    stats[bg*2+1] = rsqrtf(var + 1e-5f);
  }
}

// GN apply + relu + 1x1 conv (H->1): block = pixel, thread = channel
__global__ __launch_bounds__(128) void k_out(
    const float* __restrict__ e, const float* __restrict__ stats,
    const float* __restrict__ gamma, const float* __restrict__ beta,
    const float* __restrict__ wout, const float* __restrict__ bout,
    float* __restrict__ out)
{
  __shared__ float red[128];
  const long p = blockIdx.x;
  const int  h = threadIdx.x;
  const int  b = (int)(p >> 16);
  const int  g = h >> 2;
  float mean = stats[(b*32+g)*2], rstd = stats[(b*32+g)*2+1];
  float v = (e[p*HH + h] - mean) * rstd * gamma[h] + beta[h];
  v = fmaxf(v, 0.f) * wout[h];
  red[h] = v; __syncthreads();
  for (int s = 64; s > 0; s >>= 1) { if (h < s) red[h] += red[h+s]; __syncthreads(); }
  if (h == 0) out[p] = red[0] + bout[0];
}

// ---------------------------------------------------------------------------
extern "C" void kernel_launch(void* const* d_in, const int* in_sizes, int n_in,
                              void* d_out, int out_size, void* d_ws, size_t ws_size,
                              hipStream_t stream)
{
  (void)in_sizes; (void)n_in; (void)out_size; (void)ws_size;
  const float* label = (const float*)d_in[0];
  const float* edgew = (const float*)d_in[1];
  const int*   adj   = (const int*)  d_in[2];
  const float* xt    = (const float*)d_in[3];
  const int*   tt    = (const int*)  d_in[4];
  auto P = [&](int i) { return (const float*)d_in[i]; };

  float* ws = (float*)d_ws;
  float* e   = ws + 0*EN;
  float* ew  = ws + 1*EN;
  float* t0  = ws + 2*EN;
  float* t1  = ws + 3*EN;
  float* t2  = ws + 4*EN;
  float* sm  = ws + 5*EN;
  float* x   = sm;
  float* Uh  = sm + 1*(BV*HH);
  float* Vh  = sm + 2*(BV*HH);
  float* Ah  = sm + 3*(BV*HH);
  float* Bh  = sm + 4*(BV*HH);
  float* h_n = sm + 5*(BV*HH);
  float* tm2   = sm + 6*(BV*HH);
  float* tproj = tm2 + BB*TT;
  float* gnst  = tproj + BB*HH;

  const dim3 blk256(256), blk128(128);
  const int gE  = PIX / 128;                 // 1024 blocks (128-row GEMM)
  const int gG  = PIX / 64;                  // 2048 blocks (gate GEMM)
  const int gN  = BV / 128;                  // 4 blocks (node GEMM)
  const long nE = EN;
  const int gew = (int)((nE + 255) / 256);

  // ---- embeddings --------------------------------------------------------
  k_posembed<<<(BV*HH + 255)/256, blk256, 0, stream>>>(label, P(75), P(76), h_n, (long)BV*HH);
  gemm128_wmma<<<gN, blk256, 0, stream>>>(h_n, P(69), P(70), nullptr, x, BV);     // node_embed
  k_posembed<<<gew, blk256, 0, stream>>>(xt,    P(7),  P(8),  t0, nE);
  gemm128_wmma<<<gE, blk256, 0, stream>>>(t0, P(5),  P(6),  nullptr, e,  PIX);    // edge_embed
  k_posembed<<<gew, blk256, 0, stream>>>(edgew, P(11), P(12), t0, nE);
  gemm128_wmma<<<gE, blk256, 0, stream>>>(t0, P(9),  P(10), nullptr, ew, PIX);    // ew_embed
  k_time<<<1, blk128, 0, stream>>>(tt, P(77), P(78), P(79), P(80), tm2);

  // ---- layers ------------------------------------------------------------
  for (int l = 0; l < 2; ++l) {
    const int LB = 13 + 28*l;
    gemm128_wmma<<<gN, blk256, 0, stream>>>(x, P(LB+8),  P(LB+9),  nullptr, Uh, BV);  // U
    gemm128_wmma<<<gN, blk256, 0, stream>>>(x, P(LB+12), P(LB+13), nullptr, Vh, BV);  // V
    gemm128_wmma<<<gN, blk256, 0, stream>>>(x, P(LB+0),  P(LB+1),  nullptr, Ah, BV);  // A
    gemm128_wmma<<<gN, blk256, 0, stream>>>(x, P(LB+2),  P(LB+3),  nullptr, Bh, BV);  // B
    k_copy<<<(BV*HH + 255)/256, blk256, 0, stream>>>(h_n, Uh, (long)BV*HH);
    k_tproj<<<1, blk256, 0, stream>>>(tm2, P(LB+26), P(LB+27), tproj);

    // fused Ce/Dew/Uew/Vew GEMMs + gating + j-sum  -> t0=e_n, t1=ewn+Uew, h_n
    gemm_gate<<<gG, blk256, 0, stream>>>(
        e, ew,
        P(LB+4), P(LB+5), P(LB+6), P(LB+7), P(LB+10), P(LB+11), P(LB+14), P(LB+15),
        adj, Ah, Bh, Vh, t0, t1, h_n);

    // x = x + relu(ln_h(h_n))
    k_ln_res<<<BV, blk128, 0, stream>>>(h_n, P(LB+20), P(LB+21), x, x);
    // ew = ew + relu(ln_ew(ewn + Uew))
    k_ln_res<<<PIX, blk128, 0, stream>>>(t1, P(LB+18), P(LB+19), ew, ew);
    // t2 = silu(plo_ln(relu(ln_e(e_n)) + tproj))
    k_ln2<<<PIX, blk128, 0, stream>>>(t0, P(LB+16), P(LB+17), tproj,
                                      P(LB+24), P(LB+25), t2, PPB);
    // e = e + t2 @ plo_lin^T + b   (residual fused into epilogue, in-place)
    gemm128_wmma<<<gE, blk256, 0, stream>>>(t2, P(LB+22), P(LB+23), e, e, PIX);
  }

  // ---- output head -------------------------------------------------------
  k_gnstats<<<BB*32, blk256, 0, stream>>>(e, gnst);
  k_out<<<PIX, blk128, 0, stream>>>(e, gnst, P(73), P(74), P(71), P(72), (float*)d_out);
}
